// AutoencoderVQ_85598698209615
// MI455X (gfx1250) — matrix operand order repaired and checked by reading, not compiled, not run on hardware
//
#include <hip/hip_runtime.h>
#include <hip/hip_bf16.h>

typedef __attribute__((ext_vector_type(16))) __bf16 v16bf;
typedef __attribute__((ext_vector_type(8)))  float  v8f;
typedef __attribute__((ext_vector_type(4)))  int    v4i;

#define AS_GLOBAL __attribute__((address_space(1)))
#define AS_LDS    __attribute__((address_space(3)))

#define N_TOK 131072   // B*H*W = 32*64*64
#define HW    4096     // H*W
#define CCH   32       // latent channels C
#define DQ    32       // quant dim D
#define KCB   512      // codebook size K
#define CBS   40       // padded LDS row stride (bf16 elems): 80B, 16B-aligned, anti-bank-conflict

// ---------------------------------------------------------------------------
// Kernel 0: codebook -> bf16 copy (packed 16B stores) + per-code squared norms
// ---------------------------------------------------------------------------
__global__ void vq_cb_prep(const float* __restrict__ cb,
                           __bf16* __restrict__ cbb,
                           float* __restrict__ cn) {
    int k = blockIdx.x * blockDim.x + threadIdx.x;
    if (k >= KCB) return;
    const float* row = cb + k * DQ;
    union { __bf16 h[DQ]; uint4 u[4]; } pk;
    float s = 0.f;
#pragma unroll
    for (int d = 0; d < DQ; ++d) {
        float v = row[d];
        s += v * v;
        pk.h[d] = (__bf16)v;
    }
    uint4* dst = (uint4*)(cbb + (size_t)k * DQ);
#pragma unroll
    for (int j = 0; j < 4; ++j) dst[j] = pk.u[j];
    cn[k] = s;
}

// ---------------------------------------------------------------------------
// Kernel 1: quant 1x1 conv (fp32 math). Stores bf16(-2*xf): the -2 scale is
// exact in bf16 (sign + exponent shift) and lets the distance WMMA produce
// the score  |c|^2 - 2 x.c  directly in its accumulator.
// ---------------------------------------------------------------------------
__global__ void vq_quant_conv(const float* __restrict__ z,
                              const float* __restrict__ qw,
                              const float* __restrict__ qb,
                              __bf16* __restrict__ xfb) {
    __shared__ float w[DQ * CCH];
    __shared__ float bvec[DQ];
    for (int i = threadIdx.x; i < DQ * CCH; i += blockDim.x) w[i] = qw[i];
    if (threadIdx.x < DQ) bvec[threadIdx.x] = qb[threadIdx.x];
    __syncthreads();

    int n  = blockIdx.x * blockDim.x + threadIdx.x;   // token id
    int b  = n >> 12;
    int hw = n & (HW - 1);
    const float* zb = z + (size_t)b * CCH * HW + hw;  // coalesced across lanes
    float zv[CCH];
#pragma unroll
    for (int c = 0; c < CCH; ++c) zv[c] = zb[c * HW];

    union { __bf16 h[DQ]; uint4 u[4]; } pk;
#pragma unroll
    for (int d = 0; d < DQ; ++d) {
        float a = bvec[d];
#pragma unroll
        for (int c = 0; c < CCH; ++c) a += zv[c] * w[d * CCH + c];
        pk.h[d] = (__bf16)(-2.0f * a);
    }
    uint4* dst = (uint4*)(xfb + (size_t)n * DQ);
#pragma unroll
    for (int j = 0; j < 4; ++j) dst[j] = pk.u[j];
}

// ---------------------------------------------------------------------------
// Kernel 2: scores via v_wmma_f32_16x16x32_bf16 with |c|^2 preloaded in the
// accumulator, software-pipelined so wmma(t+1) issues before the argmin of
// tile t (fills the WMMA->VALU hazard slots with useful work).
//   block = 256 threads = 8 waves; each wave: 16 tokens vs all 512 codes.
// ---------------------------------------------------------------------------
__global__ void vq_argmin_wmma(const __bf16* __restrict__ xfb,
                               const __bf16* __restrict__ cbb,
                               const float* __restrict__ cn,
                               int* __restrict__ idxOut) {
    __shared__ __bf16 cbLds[KCB * CBS];  // 40 KB, padded rows
    __shared__ float  cnLds[KCB];        // 2 KB

#if __has_builtin(__builtin_amdgcn_global_load_async_to_lds_b128)
    // Async DMA-style staging: 2048 x 16B chunks, tracked on ASYNCcnt.
    for (int i = threadIdx.x; i < KCB * 4; i += blockDim.x) {
        int row = i >> 2, ch = i & 3;
        __builtin_amdgcn_global_load_async_to_lds_b128(
            (AS_GLOBAL v4i*)(cbb + row * DQ + ch * 8),
            (AS_LDS v4i*)(&cbLds[row * CBS + ch * 8]),
            0, 0);
    }
    for (int i = threadIdx.x; i < KCB; i += blockDim.x) cnLds[i] = cn[i];
#if __has_builtin(__builtin_amdgcn_s_wait_asynccnt)
    __builtin_amdgcn_s_wait_asynccnt(0);
#else
    asm volatile("s_wait_asynccnt 0x0" ::: "memory");
#endif
#else
    // Fallback: cooperative 16B-vector stage into padded LDS layout.
    for (int i = threadIdx.x; i < KCB * 4; i += blockDim.x) {
        int row = i >> 2, ch = i & 3;
        *(uint4*)(&cbLds[row * CBS + ch * 8]) =
            *(const uint4*)(cbb + row * DQ + ch * 8);
    }
    for (int i = threadIdx.x; i < KCB; i += blockDim.x) cnLds[i] = cn[i];
#endif
    __syncthreads();

    const int lane  = threadIdx.x & 31;
    const int wave  = threadIdx.x >> 5;
    const int T     = (blockIdx.x * 8 + wave) * 16;   // token tile base
    const int l15   = lane & 15;
    const int khalf = lane >> 4;

    // A fragment (16x32 bf16, ISA layout): lane = row (mod 16);
    // elems 0..7 -> K = khalf*8 + j ; elems 8..15 -> K = 16 + khalf*8 + (j-8)
    union { v16bf v; uint4 u[2]; } A;
    const __bf16* arow = xfb + (size_t)(T + l15) * DQ;
    A.u[0] = *(const uint4*)(arow + khalf * 8);
    A.u[1] = *(const uint4*)(arow + 16 + khalf * 8);

    float bestv[8];
    int   besti[8];
#pragma unroll
    for (int r = 0; r < 8; ++r) { bestv[r] = 3.0e38f; besti[r] = 0; }

    // Prologue: tile 0 score = (-2x).c + |c|^2 straight from the accumulator.
    v8f s_cur;
    {
        union { v16bf v; uint4 u[2]; } Bf;
        const __bf16* brow = cbLds + l15 * CBS + khalf * 8;
        Bf.u[0] = *(const uint4*)(brow);
        Bf.u[1] = *(const uint4*)(brow + 16);
        float cnv = cnLds[l15];
        v8f cini;
#pragma unroll
        for (int r = 0; r < 8; ++r) cini[r] = cnv;
        s_cur = __builtin_amdgcn_wmma_f32_16x16x32_bf16(
            false, A.v, false, Bf.v, (short)0, cini, false, false);
    }

#pragma unroll
    for (int t = 0; t < KCB / 16; ++t) {
        v8f s_nxt = s_cur;
        if (t + 1 < KCB / 16) {
            // Issue next tile's B loads + WMMA *before* consuming s_cur:
            // its hazard window is covered by the argmin VALU below.
            union { v16bf v; uint4 u[2]; } Bf;
            const __bf16* brow = cbLds + ((t + 1) * 16 + l15) * CBS + khalf * 8;
            Bf.u[0] = *(const uint4*)(brow);
            Bf.u[1] = *(const uint4*)(brow + 16);
            float cnv = cnLds[(t + 1) * 16 + l15];
            v8f cini;
#pragma unroll
            for (int r = 0; r < 8; ++r) cini[r] = cnv;
            s_nxt = __builtin_amdgcn_wmma_f32_16x16x32_bf16(
                false, A.v, false, Bf.v, (short)0, cini, false, false);
        }
        const int cand = t * 16 + l15;
#pragma unroll
        for (int r = 0; r < 8; ++r) {
            float sc = s_cur[r];
            if (sc < bestv[r]) { bestv[r] = sc; besti[r] = cand; }
        }
        s_cur = s_nxt;
    }

    // C layout: lanes 0-15 hold rows 0..7 (per acc VGPR r), lanes 16-31 rows 8..15.
    // Reduce across the 16 lanes of each half; tie-break on lower index.
#pragma unroll
    for (int r = 0; r < 8; ++r) {
#pragma unroll
        for (int off = 8; off > 0; off >>= 1) {
            float ov = __shfl_xor(bestv[r], off, 32);
            int   oi = __shfl_xor(besti[r], off, 32);
            if (ov < bestv[r] || (ov == bestv[r] && oi < besti[r])) {
                bestv[r] = ov; besti[r] = oi;
            }
        }
    }
    if (l15 == 0) {
#pragma unroll
        for (int r = 0; r < 8; ++r) idxOut[T + khalf * 8 + r] = besti[r];
    }
}

// ---------------------------------------------------------------------------
// Kernel 3: gather q, post_quant conv -> out, per-block loss partials (fp32)
// ---------------------------------------------------------------------------
__global__ void vq_out_loss(const float* __restrict__ z,
                            const float* __restrict__ qw,
                            const float* __restrict__ qb,
                            const float* __restrict__ cb,
                            const float* __restrict__ pqw,
                            const float* __restrict__ pqb,
                            const int* __restrict__ idxIn,
                            float* __restrict__ outp,
                            float* __restrict__ partials) {
    __shared__ float w[DQ * CCH];
    __shared__ float pw[CCH * DQ];
    __shared__ float bq[DQ];
    __shared__ float bp[CCH];
    __shared__ float red[256];
    for (int i = threadIdx.x; i < DQ * CCH; i += blockDim.x) w[i]  = qw[i];
    for (int i = threadIdx.x; i < CCH * DQ; i += blockDim.x) pw[i] = pqw[i];
    if (threadIdx.x < DQ)  bq[threadIdx.x] = qb[threadIdx.x];
    if (threadIdx.x < CCH) bp[threadIdx.x] = pqb[threadIdx.x];
    __syncthreads();

    int n  = blockIdx.x * blockDim.x + threadIdx.x;
    int b  = n >> 12;
    int hw = n & (HW - 1);
    const float* zb = z + (size_t)b * CCH * HW + hw;
    float zv[CCH];
#pragma unroll
    for (int c = 0; c < CCH; ++c) zv[c] = zb[c * HW];

    int k = idxIn[n];
    float q[DQ];
#pragma unroll
    for (int d = 0; d < DQ; ++d) q[d] = cb[k * DQ + d];

    // recompute xf in fp32 for the loss (exact-precision path)
    float ls = 0.f;
#pragma unroll
    for (int d = 0; d < DQ; ++d) {
        float xf = bq[d];
#pragma unroll
        for (int c = 0; c < CCH; ++c) xf += zv[c] * w[d * CCH + c];
        float dd = q[d] - xf;
        ls += dd * dd;
    }

    // out = q @ post_quant_w^T + b  (q_st == q numerically)
    float* ob = outp + (size_t)b * CCH * HW + hw;
#pragma unroll
    for (int c = 0; c < CCH; ++c) {
        float a = bp[c];
#pragma unroll
        for (int d = 0; d < DQ; ++d) a += q[d] * pw[c * DQ + d];
        ob[c * HW] = a;
    }

    // deterministic block tree-reduction of loss partials
    red[threadIdx.x] = ls;
    __syncthreads();
    for (int sft = 128; sft > 0; sft >>= 1) {
        if ((int)threadIdx.x < sft) red[threadIdx.x] += red[threadIdx.x + sft];
        __syncthreads();
    }
    if (threadIdx.x == 0) partials[blockIdx.x] = red[0];
}

// ---------------------------------------------------------------------------
// Kernel 4: deterministic final loss reduction
// ---------------------------------------------------------------------------
__global__ void vq_loss_final(const float* __restrict__ partials,
                              float* __restrict__ lossOut, int nparts) {
    if (blockIdx.x == 0 && threadIdx.x == 0) {
        float s = 0.f;
        for (int i = 0; i < nparts; ++i) s += partials[i];
        *lossOut = 2.0f * s / (float)(N_TOK * DQ);  // commit + codebook loss
    }
}

// ---------------------------------------------------------------------------
extern "C" void kernel_launch(void* const* d_in, const int* in_sizes, int n_in,
                              void* d_out, int out_size, void* d_ws, size_t ws_size,
                              hipStream_t stream) {
    const float* z   = (const float*)d_in[0];
    const float* qw  = (const float*)d_in[1];
    const float* qb  = (const float*)d_in[2];
    const float* cb  = (const float*)d_in[3];
    const float* pqw = (const float*)d_in[4];
    const float* pqb = (const float*)d_in[5];

    float* out     = (float*)d_out;                                       // [32,32,64,64]
    int*   idxOut  = (int*)((char*)d_out + (size_t)N_TOK * CCH * 4);      // [32,4096] int32
    float* lossOut = (float*)((char*)d_out + ((size_t)N_TOK * CCH + N_TOK) * 4);

    char*   ws       = (char*)d_ws;
    __bf16* xfb      = (__bf16*)ws;                                        // 8 MB
    __bf16* cbb      = (__bf16*)(ws + (size_t)N_TOK * DQ * 2);             // 32 KB
    float*  cn       = (float*)(ws + (size_t)N_TOK * DQ * 2 + KCB * DQ * 2);
    float*  partials = cn + KCB;                                           // 512 floats

    vq_cb_prep    <<<(KCB + 255) / 256, 256, 0, stream>>>(cb, cbb, cn);
    vq_quant_conv <<<N_TOK / 256,       256, 0, stream>>>(z, qw, qb, xfb);
    vq_argmin_wmma<<<N_TOK / 128,       256, 0, stream>>>(xfb, cbb, cn, idxOut);
    vq_out_loss   <<<N_TOK / 256,       256, 0, stream>>>(z, qw, qb, cb, pqw, pqb,
                                                          idxOut, out, partials);
    vq_loss_final <<<1, 32, 0, stream>>>(partials, lossOut, N_TOK / 256);
}
